// DBGCN_74380243632206
// MI455X (gfx1250) — compile-verified, matrix operand order checked by reading, not run on hardware
//
#include <hip/hip_runtime.h>
#include <hip/hip_bf16.h>
#include <cmath>

#define NFEAT   512
#define NHID    256
#define NCLASS  40
#define NLAYERS 8

typedef __attribute__((ext_vector_type(16))) __bf16 v16bf;
typedef __attribute__((ext_vector_type(8)))  __bf16 v8bf;
typedef __attribute__((ext_vector_type(8)))  float  v8f;

__device__ __forceinline__ __bf16 f2bf(float f) { return (__bf16)f; }

// Load a 16x32 bf16 fragment row (A) or column (B from pre-transposed Bt).
// Per CDNA5 ISA 16-bit A layout: lanes 0-15 hold K={0..7,16..23}, lanes 16-31
// hold K={8..15,24..31}; caller bakes the (lane>>4)*8 offset into `p`.
__device__ __forceinline__ v16bf load_frag16(const __bf16* p) {
    v8bf lo = *(const v8bf*)p;          // K + {0..7}
    v8bf hi = *(const v8bf*)(p + 16);   // K + {16..23}
    return __builtin_shufflevector(lo, hi, 0,1,2,3,4,5,6,7,8,9,10,11,12,13,14,15);
}

// ---------------------------------------------------------------------------
// WMMA GEMM: C[M,256] = A(bf16)[M,K] x B(bf16)[K,256], Bt stored as [256][K].
// Block = 256 threads = 8 waves; wave tile = 32 rows x 64 cols (8 wmma accs).
// B tile (64 cols x K, shared by all 8 waves) is staged once in LDS with a
// +8-element pad (stride 260 dwords -> conflict-free ds_load_b128 fragments).
// A fragments stream from global with a 2x-unrolled register ping-pong so no
// cross-iteration register copies are needed and loads stay in flight.
// mode 0: FC    : v = relu(acc + bias)          -> h0f, hf, supbf[:,256:], dbf
// mode 1: CONV  : v = relu(th*acc + (1-th)*(0.9*hi+0.1*h0) + h) -> hf
// mode 2: DYN   : v = relu(acc + bias) + res*h0 -> d32, dbf
// ---------------------------------------------------------------------------
__global__ __launch_bounds__(256) void gemm_bf16_wmma(
    const __bf16* __restrict__ A, int lda,
    const __bf16* __restrict__ Bt, int K, int M,
    int mode, float theta, float resScale,
    const float*  __restrict__ bias,
    float*        __restrict__ h0f,
    const float*  __restrict__ hif,
    float*        __restrict__ hf,
    float*        __restrict__ d32,
    __bf16*       __restrict__ dbf,
    __bf16*       __restrict__ supbf)
{
    extern __shared__ __bf16 sB[];           // 64 * (K+8) bf16
    const int lane   = threadIdx.x & 31;
    const int wave   = threadIdx.x >> 5;
    const int row0   = blockIdx.x * 256 + wave * 32;
    const int col0   = blockIdx.y * 64;
    const int khalf  = (lane >> 4) << 3;     // 0 or 8
    const int strB   = K + 8;                // padded LDS col stride (elements)

    // ---- cooperative stage of the 64 x K B tile into LDS ----
    {
        const int chunksPerCol = K >> 3;                 // 16B chunks per col
        const int totalChunks  = 64 * chunksPerCol;
        for (int cc = threadIdx.x; cc < totalChunks; cc += 256) {
            int c  = cc / chunksPerCol;
            int kc = (cc - c * chunksPerCol) << 3;
            v8bf v = *(const v8bf*)(Bt + (size_t)(col0 + c) * K + kc);
            *(v8bf*)(sB + c * strB + kc) = v;
        }
    }
    __syncthreads();

    int arow0 = row0 + (lane & 15);
    int arow1 = row0 + 16 + (lane & 15);
    if (arow0 > M - 1) arow0 = M - 1;        // clamp; OOB rows masked at store
    if (arow1 > M - 1) arow1 = M - 1;
    const __bf16* aB0 = A + (size_t)arow0 * lda + khalf;
    const __bf16* aB1 = A + (size_t)arow1 * lda + khalf;

    int boff[4];
#pragma unroll
    for (int nt = 0; nt < 4; ++nt)
        boff[nt] = (nt * 16 + (lane & 15)) * strB + khalf;

    v8f acc[2][4];
#pragma unroll
    for (int h = 0; h < 2; ++h)
#pragma unroll
        for (int nt = 0; nt < 4; ++nt) {
            v8f z = {0.f,0.f,0.f,0.f,0.f,0.f,0.f,0.f};
            acc[h][nt] = z;
        }

    // one 32-deep k-step: 4 B fragments from LDS, 8 WMMAs
    auto mma_step = [&](int k, const v16bf& a0, const v16bf& a1) {
#pragma unroll
        for (int nt = 0; nt < 4; ++nt) {
            v16bf b = load_frag16(sB + boff[nt] + k);
            acc[0][nt] = __builtin_amdgcn_wmma_f32_16x16x32_bf16(
                false, a0, false, b, (short)0, acc[0][nt], false, false);
            acc[1][nt] = __builtin_amdgcn_wmma_f32_16x16x32_bf16(
                false, a1, false, b, (short)0, acc[1][nt], false, false);
        }
    };

    // ---- 2x-unrolled ping-pong over K (K is a multiple of 64) ----
    v16bf pa0 = load_frag16(aB0), pa1 = load_frag16(aB1);
    const int KK = K >> 6;
    for (int it = 0; it < KK - 1; ++it) {
        const int k = it << 6;
        v16bf qa0 = load_frag16(aB0 + k + 32);
        v16bf qa1 = load_frag16(aB1 + k + 32);
        mma_step(k, pa0, pa1);
        pa0 = load_frag16(aB0 + k + 64);
        pa1 = load_frag16(aB1 + k + 64);
        mma_step(k + 32, qa0, qa1);
    }
    {
        v16bf qa0 = load_frag16(aB0 + K - 32);
        v16bf qa1 = load_frag16(aB1 + K - 32);
        mma_step(K - 64, pa0, pa1);
        mma_step(K - 32, qa0, qa1);
    }

    // ---- fused epilogues ----
    const float onemtheta = 1.0f - theta;
#pragma unroll
    for (int half = 0; half < 2; ++half) {
#pragma unroll
        for (int nt = 0; nt < 4; ++nt) {
            const int col = col0 + nt * 16 + (lane & 15);
#pragma unroll
            for (int r = 0; r < 8; ++r) {
                const int m = row0 + half * 16 + khalf + r;
                if (m >= M) continue;
                const size_t idx = (size_t)m * NHID + col;
                float v = acc[half][nt][r];
                if (mode == 0) {
                    v += bias[col];
                    v = v > 0.f ? v : 0.f;
                    h0f[idx] = v;
                    hf[idx]  = v;
                    supbf[(size_t)m * (2 * NHID) + NHID + col] = f2bf(v);
                    dbf[idx] = f2bf(v);
                } else if (mode == 1) {
                    float rr = 0.9f * hif[idx] + 0.1f * h0f[idx];
                    v = theta * v + onemtheta * rr + hf[idx];
                    v = v > 0.f ? v : 0.f;
                    hf[idx] = v;
                } else {
                    v += bias[col];
                    v = v > 0.f ? v : 0.f;
                    v += resScale * h0f[idx];
                    d32[idx] = v;
                    dbf[idx] = f2bf(v);
                }
            }
        }
    }
}

// ---------------------------------------------------------------------------
// Weight prep / conversion kernels
// ---------------------------------------------------------------------------
__global__ void transpose_f32_to_bf16(const float* __restrict__ src,
                                      __bf16* __restrict__ dst,
                                      int K, int Ncol, long long total) {
    long long i = (long long)blockIdx.x * blockDim.x + threadIdx.x;
    if (i >= total) return;
    const long long per = (long long)K * Ncol;
    long long b = i / per, rem = i % per;
    int k = (int)(rem / Ncol), n = (int)(rem % Ncol);
    dst[(size_t)b * per + (size_t)n * K + k] = f2bf(src[i]);
}

// dyn_eff_t[l][n][k] = dyn_w[l][k][n] - dyn_w[l][256+k][n]  (cat([d,-d]) fold)
__global__ void make_dyneff(const float* __restrict__ dynw,
                            __bf16* __restrict__ dst, long long total) {
    long long i = (long long)blockIdx.x * blockDim.x + threadIdx.x;
    if (i >= total) return;
    int l = (int)(i / (NHID * NHID));
    int rem = (int)(i % (NHID * NHID));
    int n = rem / NHID, k = rem % NHID;
    const float* base = dynw + (size_t)l * (2 * NHID) * NHID;
    float v = base[(size_t)k * NHID + n] - base[(size_t)(k + NHID) * NHID + n];
    dst[(size_t)l * NHID * NHID + (size_t)n * NHID + k] = f2bf(v);
}

__global__ void cvt_f32_bf16(const float* __restrict__ src,
                             __bf16* __restrict__ dst, long long n) {
    long long i = (long long)blockIdx.x * blockDim.x + threadIdx.x;
    if (i < n) dst[i] = f2bf(src[i]);
}

__global__ void hi_to_support(const float* __restrict__ hi,
                              __bf16* __restrict__ sup, long long total) {
    long long i = (long long)blockIdx.x * blockDim.x + threadIdx.x;
    if (i >= total) return;
    long long row = i >> 8;
    int c = (int)(i & 255);
    sup[(size_t)row * (2 * NHID) + c] = f2bf(hi[i]);
}

__global__ void fill_zero(float* __restrict__ p, long long n) {
    long long i = (long long)blockIdx.x * blockDim.x + threadIdx.x;
    if (i < n) p[i] = 0.f;
}

// ---------------------------------------------------------------------------
// SpMM (scatter-atomic): hi[dst] += w * h[src]; one wave per edge, one lane
// per 8-feature chunk. h (102MB) is L2-resident (192MB L2), so the random
// gathers and atomics stay on-chip.
// ---------------------------------------------------------------------------
__global__ __launch_bounds__(256) void spmm_atomic(
    const int* __restrict__ esrc, const int* __restrict__ edst,
    const float* __restrict__ ew, const float* __restrict__ h,
    float* __restrict__ hi, long long E)
{
    long long t = (long long)blockIdx.x * 256 + threadIdx.x;
    long long e = t >> 5;
    if (e >= E) return;
    int ch = (int)(t & 31);
    int s = esrc[e], d = edst[e];
    float wt = ew[e];
    const float4* hp = (const float4*)(h + (size_t)s * NHID + ch * 8);
    float4 a = hp[0], b = hp[1];
    float* o = hi + (size_t)d * NHID + ch * 8;
    __hip_atomic_fetch_add(o + 0, wt * a.x, __ATOMIC_RELAXED, __HIP_MEMORY_SCOPE_AGENT);
    __hip_atomic_fetch_add(o + 1, wt * a.y, __ATOMIC_RELAXED, __HIP_MEMORY_SCOPE_AGENT);
    __hip_atomic_fetch_add(o + 2, wt * a.z, __ATOMIC_RELAXED, __HIP_MEMORY_SCOPE_AGENT);
    __hip_atomic_fetch_add(o + 3, wt * a.w, __ATOMIC_RELAXED, __HIP_MEMORY_SCOPE_AGENT);
    __hip_atomic_fetch_add(o + 4, wt * b.x, __ATOMIC_RELAXED, __HIP_MEMORY_SCOPE_AGENT);
    __hip_atomic_fetch_add(o + 5, wt * b.y, __ATOMIC_RELAXED, __HIP_MEMORY_SCOPE_AGENT);
    __hip_atomic_fetch_add(o + 6, wt * b.z, __ATOMIC_RELAXED, __HIP_MEMORY_SCOPE_AGENT);
    __hip_atomic_fetch_add(o + 7, wt * b.w, __ATOMIC_RELAXED, __HIP_MEMORY_SCOPE_AGENT);
}

// ---------------------------------------------------------------------------
// cross = LN(h)*g1+b1 + LN(d)*g2+b2. One wave32 per row (8 feats/lane).
// ---------------------------------------------------------------------------
__global__ __launch_bounds__(256) void ln_cross(
    const float* __restrict__ h, const float* __restrict__ d,
    const float* __restrict__ g1, const float* __restrict__ b1,
    const float* __restrict__ g2, const float* __restrict__ b2,
    float* __restrict__ cross, int N)
{
    const int lane = threadIdx.x & 31, wave = threadIdx.x >> 5;
    const int row = blockIdx.x * 8 + wave;
    if (row >= N) return;
    const float4* hp = (const float4*)(h + (size_t)row * NHID + lane * 8);
    const float4* dp = (const float4*)(d + (size_t)row * NHID + lane * 8);
    float4 h0v = hp[0], h1v = hp[1], d0v = dp[0], d1v = dp[1];
    float hv[8] = {h0v.x,h0v.y,h0v.z,h0v.w,h1v.x,h1v.y,h1v.z,h1v.w};
    float dv[8] = {d0v.x,d0v.y,d0v.z,d0v.w,d1v.x,d1v.y,d1v.z,d1v.w};

    float sh = 0.f, sd = 0.f;
#pragma unroll
    for (int i = 0; i < 8; ++i) { sh += hv[i]; sd += dv[i]; }
#pragma unroll
    for (int off = 16; off > 0; off >>= 1) {
        sh += __shfl_xor(sh, off, 32);
        sd += __shfl_xor(sd, off, 32);
    }
    const float mh = sh * (1.0f / NHID), md = sd * (1.0f / NHID);

    float vh = 0.f, vd = 0.f;
#pragma unroll
    for (int i = 0; i < 8; ++i) {
        float a = hv[i] - mh, c = dv[i] - md;
        vh += a * a; vd += c * c;
    }
#pragma unroll
    for (int off = 16; off > 0; off >>= 1) {
        vh += __shfl_xor(vh, off, 32);
        vd += __shfl_xor(vd, off, 32);
    }
    const float inv1 = rsqrtf(vh * (1.0f / NHID) + 1e-6f);
    const float inv2 = rsqrtf(vd * (1.0f / NHID) + 1e-6f);

    float* outr = cross + (size_t)row * NHID;
#pragma unroll
    for (int i = 0; i < 8; ++i) {
        int c = lane * 8 + i;
        outr[c] = g1[c] * (hv[i] - mh) * inv1 + b1[c]
                + g2[c] * (dv[i] - md) * inv2 + b2[c];
    }
}

// ---------------------------------------------------------------------------
// logits = cross @ W_out + b_out; log_softmax. Block(64) per node.
// ---------------------------------------------------------------------------
__global__ __launch_bounds__(64) void out_logsoftmax(
    const float* __restrict__ cross, const float* __restrict__ Wout,
    const float* __restrict__ bout, float* __restrict__ out, int N)
{
    const int row = blockIdx.x;
    if (row >= N) return;
    __shared__ float rs[NHID];
    __shared__ float lg[NCLASS];
    for (int i = threadIdx.x; i < NHID; i += 64)
        rs[i] = cross[(size_t)row * NHID + i];
    __syncthreads();
    const int c = threadIdx.x;
    if (c < NCLASS) {
        float s = bout[c];
        for (int k = 0; k < NHID; ++k) s += rs[k] * Wout[(size_t)k * NCLASS + c];
        lg[c] = s;
    }
    __syncthreads();
    if (c < NCLASS) {
        float mx = -3.4e38f;
        for (int j = 0; j < NCLASS; ++j) mx = fmaxf(mx, lg[j]);
        float se = 0.f;
        for (int j = 0; j < NCLASS; ++j) se += expf(lg[j] - mx);
        out[(size_t)row * NCLASS + c] = lg[c] - mx - logf(se);
    }
}

// ---------------------------------------------------------------------------
extern "C" void kernel_launch(void* const* d_in, const int* in_sizes, int n_in,
                              void* d_out, int out_size, void* d_ws, size_t ws_size,
                              hipStream_t stream) {
    const float* x     = (const float*)d_in[0];
    const int*   esrc  = (const int*)  d_in[1];
    const int*   edst  = (const int*)  d_in[2];
    const float* ew    = (const float*)d_in[3];
    const float* Wfc   = (const float*)d_in[4];
    const float* bfc   = (const float*)d_in[5];
    const float* convw = (const float*)d_in[6];
    const float* dynw  = (const float*)d_in[7];
    const float* dynb  = (const float*)d_in[8];
    const float* g1    = (const float*)d_in[9];
    const float* be1   = (const float*)d_in[10];
    const float* g2    = (const float*)d_in[11];
    const float* be2   = (const float*)d_in[12];
    const float* Wout  = (const float*)d_in[13];
    const float* bout  = (const float*)d_in[14];

    const int       N = in_sizes[0] / NFEAT;
    const long long E = in_sizes[1];

    // ---- workspace carving (~666 MB) ----
    char* p = (char*)d_ws;
    auto carve = [&](size_t bytes) -> void* {
        void* q = (void*)p;
        p += (bytes + 255) & ~(size_t)255;
        return q;
    };
    __bf16* WfcT  = (__bf16*)carve((size_t)NHID * NFEAT * 2);
    __bf16* ConvT = (__bf16*)carve((size_t)NLAYERS * NHID * NFEAT * 2);
    __bf16* DynT  = (__bf16*)carve((size_t)NLAYERS * NHID * NHID * 2);
    __bf16* Xbf   = (__bf16*)carve((size_t)N * NFEAT * 2);
    __bf16* Sup   = (__bf16*)carve((size_t)N * (2 * NHID) * 2);
    float*  H0    = (float*) carve((size_t)N * NHID * 4);
    float*  H     = (float*) carve((size_t)N * NHID * 4);
    float*  HI    = (float*) carve((size_t)N * NHID * 4);
    float*  D32   = (float*) carve((size_t)N * NHID * 4);
    __bf16* Dbf   = (__bf16*)carve((size_t)N * NHID * 2);

    float* logits = (float*)d_out;
    float* cross  = (float*)d_out + (size_t)N * NCLASS;

    const dim3 gblk(256, 1, 1);
    const dim3 ggrid((N + 255) / 256, NHID / 64, 1);
    const size_t lds512 = (size_t)64 * (NFEAT + 8) * sizeof(__bf16);  // K=512 tile
    const size_t lds256 = (size_t)64 * (NHID  + 8) * sizeof(__bf16);  // K=256 tile

    // ---- one-time weight prep ----
    {
        long long tot = (long long)NFEAT * NHID;
        transpose_f32_to_bf16<<<(tot + 255) / 256, 256, 0, stream>>>(Wfc, WfcT, NFEAT, NHID, tot);
    }
    {
        long long tot = (long long)NLAYERS * NFEAT * NHID;
        transpose_f32_to_bf16<<<(tot + 255) / 256, 256, 0, stream>>>(convw, ConvT, NFEAT, NHID, tot);
    }
    {
        long long tot = (long long)NLAYERS * NHID * NHID;
        make_dyneff<<<(tot + 255) / 256, 256, 0, stream>>>(dynw, DynT, tot);
    }
    {
        long long tot = (long long)N * NFEAT;
        cvt_f32_bf16<<<(tot + 255) / 256, 256, 0, stream>>>(x, Xbf, tot);
    }

    // ---- h0 = relu(x @ W_fc + b_fc); fan out to h, support[:,256:], d ----
    gemm_bf16_wmma<<<ggrid, gblk, lds512, stream>>>(
        Xbf, NFEAT, WfcT, NFEAT, N,
        /*mode=*/0, /*theta=*/0.f, /*res=*/0.f,
        bfc, H0, nullptr, H, nullptr, Dbf, Sup);

    // ---- 8 GCNII layers + dynamic branch ----
    const long long nh = (long long)N * NHID;
    for (int i = 0; i < NLAYERS; ++i) {
        const float theta = logf(0.5f / (float)(i + 1) + 1.0f);

        fill_zero<<<(nh + 255) / 256, 256, 0, stream>>>(HI, nh);
        {
            long long tot = E * 32;
            spmm_atomic<<<(tot + 255) / 256, 256, 0, stream>>>(esrc, edst, ew, H, HI, E);
        }
        hi_to_support<<<(nh + 255) / 256, 256, 0, stream>>>(HI, Sup, nh);

        // h = relu(theta*([hi,h0]@conv_w) + (1-theta)*(0.9*hi+0.1*h0) + h)
        gemm_bf16_wmma<<<ggrid, gblk, lds512, stream>>>(
            Sup, 2 * NHID, ConvT + (size_t)i * NHID * NFEAT, NFEAT, N,
            /*mode=*/1, theta, 0.f,
            nullptr, H0, HI, H, nullptr, nullptr, nullptr);

        // d = relu(d @ dyn_eff + dyn_b) [+ 0.1*h0 for i>0]
        gemm_bf16_wmma<<<ggrid, gblk, lds256, stream>>>(
            Dbf, NHID, DynT + (size_t)i * NHID * NHID, NHID, N,
            /*mode=*/2, 0.f, (i == 0) ? 0.f : 0.1f,
            dynb + (size_t)i * NHID, H0, nullptr, nullptr, D32, Dbf, nullptr);
    }

    // ---- cross = LN(h) + LN(d); logits + log_softmax ----
    ln_cross<<<(N + 7) / 8, 256, 0, stream>>>(H, D32, g1, be1, g2, be2, cross, N);
    out_logsoftmax<<<N, 64, 0, stream>>>(cross, Wout, bout, logits, N);
}
